// Attention_23613730193761
// MI455X (gfx1250) — compile-verified, matrix operand order checked by reading, not run on hardware
//
#include <hip/hip_runtime.h>

typedef __attribute__((ext_vector_type(16))) __bf16 v16bf;
typedef __attribute__((ext_vector_type(8)))  float  v8f;
typedef unsigned short ushort_t;

#define B_    64
#define H_    12
#define NTOK  388
#define C_    768
#define D_    64
#define NKP   416          // NTOK padded to multiple of 32
#define MTOT  (B_ * NTOK)  // 24832 = 194 * 128
#define LT    128          // temporal query count
#define LS    260          // spatial query count

// ---------- helpers ----------
__device__ inline ushort_t f2bf(float f) {
  union { float f; unsigned u; } x; x.f = f;
  unsigned r = x.u + 0x7FFFu + ((x.u >> 16) & 1u);  // RNE
  return (ushort_t)(r >> 16);
}

__device__ inline v8f wmma_bf16(v16bf a, v16bf b, v8f c) {
  return __builtin_amdgcn_wmma_f32_16x16x32_bf16(
      /*neg_a=*/false, a, /*neg_b=*/false, b,
      /*c_mod=*/(short)0, c, /*reuse_a=*/false, /*reuse_b=*/false);
}

// Load a 16x32 bf16 A/B fragment from row-major [rows, ld] at (row_base, k_base).
// ISA 16-bit A 16x32 layout: lane l<16 holds row l, K = k_base+{0..7,16..23};
// lane l+16 holds row l, K = k_base+{8..15,24..31}.  (B mirrors with N<-rows.)
__device__ inline v16bf load_frag(const ushort_t* __restrict__ src, int ld,
                                  int row_base, int k_base) {
  int lane = threadIdx.x & 31;
  int l15 = lane & 15, hi = lane >> 4;
  const ushort_t* p = src + (size_t)(row_base + l15) * ld + k_base + hi * 8;
  v16bf a;
  ((uint4*)&a)[0] = *(const uint4*)(p);
  ((uint4*)&a)[1] = *(const uint4*)(p + 16);
  return a;
}

__device__ inline float half_reduce_max(float v) {
  #pragma unroll
  for (int m = 1; m < 16; m <<= 1) v = fmaxf(v, __shfl_xor(v, m, 32));
  return v;
}
__device__ inline float half_reduce_sum(float v) {
  #pragma unroll
  for (int m = 1; m < 16; m <<= 1) v += __shfl_xor(v, m, 32);
  return v;
}

// ---------- kernel 0: f32 -> bf16 bulk convert (vectorized x4) ----------
__global__ void cvt4_kernel(const float4* __restrict__ src, ushort4* __restrict__ dst, int n4) {
  int i = blockIdx.x * blockDim.x + threadIdx.x;
  if (i < n4) {
    float4 f = src[i];
    ushort4 r;
    r.x = f2bf(f.x); r.y = f2bf(f.y); r.z = f2bf(f.z); r.w = f2bf(f.w);
    dst[i] = r;
  }
}

// ---------- kernel 0b: zero sequence padding rows (388..415) of q/k/vT ----------
__global__ void zero_pad_kernel(ushort_t* q, ushort_t* k, ushort_t* vT) {
  int i = blockIdx.x * blockDim.x + threadIdx.x;
  const int total = B_ * H_ * (NKP - NTOK) * D_;
  if (i >= total) return;
  int d = i & 63;
  int r = i >> 6;
  int t = NTOK + (r % (NKP - NTOK));
  int bh = r / (NKP - NTOK);
  q[((size_t)bh * NKP + t) * D_ + d] = 0;
  k[((size_t)bh * NKP + t) * D_ + d] = 0;
  vT[((size_t)bh * D_ + d) * NKP + t] = 0;
}

// ---------- kernel 1: QKV GEMM (x[24832,768] @ qkv_w^T[768,2304] + b) ----------
// grid (194, 18), block 256 (8 waves as 4M x 2N). Wave tile: 32M x 64N,
// software-pipelined K loop (step 32). Epilogue adds bias, folds 1/sqrt(D)
// into q, scatters bf16 into q[B,H,NKP,D], k[B,H,NKP,D], vT[B,H,D,NKP].
__global__ __launch_bounds__(256) void qkv_gemm_kernel(
    const ushort_t* __restrict__ xb, const ushort_t* __restrict__ wb,
    const float* __restrict__ bias,
    ushort_t* __restrict__ qo, ushort_t* __restrict__ ko, ushort_t* __restrict__ vo) {
  int wid = threadIdx.x >> 5;
  int lane = threadIdx.x & 31, l15 = lane & 15, hi = lane >> 4;
  int mb = blockIdx.x * 128 + (wid >> 1) * 32;
  int nb = blockIdx.y * 128 + (wid & 1) * 64;

  v8f c[2][4] = {};
  // prologue loads for K-step 0
  v16bf a0 = load_frag(xb, C_, mb,      0);
  v16bf a1 = load_frag(xb, C_, mb + 16, 0);
  v16bf b0 = load_frag(wb, C_, nb,      0);
  v16bf b1 = load_frag(wb, C_, nb + 16, 0);
  v16bf b2 = load_frag(wb, C_, nb + 32, 0);
  v16bf b3 = load_frag(wb, C_, nb + 48, 0);

  for (int ks = 0; ks < C_; ks += 32) {
    v16bf ca0 = a0, ca1 = a1, cb0 = b0, cb1 = b1, cb2 = b2, cb3 = b3;
    int nk = ks + 32;
    if (nk < C_) {  // prefetch next K-step while this step's WMMAs run
      a0 = load_frag(xb, C_, mb,      nk);
      a1 = load_frag(xb, C_, mb + 16, nk);
      b0 = load_frag(wb, C_, nb,      nk);
      b1 = load_frag(wb, C_, nb + 16, nk);
      b2 = load_frag(wb, C_, nb + 32, nk);
      b3 = load_frag(wb, C_, nb + 48, nk);
    }
    c[0][0] = wmma_bf16(ca0, cb0, c[0][0]);
    c[0][1] = wmma_bf16(ca0, cb1, c[0][1]);
    c[0][2] = wmma_bf16(ca0, cb2, c[0][2]);
    c[0][3] = wmma_bf16(ca0, cb3, c[0][3]);
    c[1][0] = wmma_bf16(ca1, cb0, c[1][0]);
    c[1][1] = wmma_bf16(ca1, cb1, c[1][1]);
    c[1][2] = wmma_bf16(ca1, cb2, c[1][2]);
    c[1][3] = wmma_bf16(ca1, cb3, c[1][3]);
  }

  #pragma unroll
  for (int s = 0; s < 2; ++s) {
    #pragma unroll
    for (int t = 0; t < 4; ++t) {
      int g = nb + t * 16 + l15;        // 0..2303
      float bv = bias[g];
      int sec = g / C_;                 // 0=q 1=k 2=v
      int cc = g - sec * C_;
      int h = cc >> 6, d = cc & 63;
      #pragma unroll
      for (int j = 0; j < 8; ++j) {
        int r = mb + s * 16 + hi * 8 + j;   // token row, < 24832
        int bb = r / NTOK, tk = r - bb * NTOK;
        float v = c[s][t][j] + bv;
        size_t bh = (size_t)bb * H_ + h;
        if (sec == 0)      qo[(bh * NKP + tk) * D_ + d] = f2bf(v * 0.125f);  // * D^-0.5
        else if (sec == 1) ko[(bh * NKP + tk) * D_ + d] = f2bf(v);
        else               vo[(bh * D_ + d) * NKP + tk] = f2bf(v);
      }
    }
  }
}

// ---------- kernel 2: flash attention (launched for temporal & spatial) ----------
// grid (ceil(q_len/64), B*H), block 128 (4 waves); each wave: 16 queries,
// online softmax over 32-key blocks. K-frags prefetched one block ahead;
// V-frags issued before the softmax VALU work to hide their latency.
__global__ __launch_bounds__(128) void flash_kernel(
    const ushort_t* __restrict__ qb, const ushort_t* __restrict__ kb,
    const ushort_t* __restrict__ vt, ushort_t* __restrict__ ao,
    int q_start, int q_len, int k_len) {
  int bh = blockIdx.y;
  int b = bh / H_, h = bh % H_;
  int wid = threadIdx.x >> 5;
  int lane = threadIdx.x & 31, l15 = lane & 15, hi = lane >> 4;
  int qbase = blockIdx.x * 64 + wid * 16;

  __shared__ __align__(16) ushort_t lds_p[4][16 * 32];
  if (qbase >= q_len) return;   // no barriers used; safe early-exit

  const ushort_t* qbh = qb + (size_t)bh * NKP * D_;
  const ushort_t* kbh = kb + (size_t)bh * NKP * D_;
  const ushort_t* vbh = vt + (size_t)bh * D_ * NKP;
  ushort_t* my = lds_p[wid];

  v16bf aq0 = load_frag(qbh, D_, q_start + qbase, 0);
  v16bf aq1 = load_frag(qbh, D_, q_start + qbase, 32);

  v8f o0 = {}, o1 = {}, o2 = {}, o3 = {};
  float mrow[8], lrow[8];
  #pragma unroll
  for (int j = 0; j < 8; ++j) { mrow[j] = -1e30f; lrow[j] = 0.0f; }

  int nkb = (k_len + 31) / 32;
  // prologue: K fragments for block 0
  v16bf k00 = load_frag(kbh, D_, 0,  0);
  v16bf k01 = load_frag(kbh, D_, 0,  32);
  v16bf k10 = load_frag(kbh, D_, 16, 0);
  v16bf k11 = load_frag(kbh, D_, 16, 32);

  for (int kblk = 0; kblk < nkb; ++kblk) {
    int kb0 = kblk * 32;
    v16bf ck00 = k00, ck01 = k01, ck10 = k10, ck11 = k11;

    // V fragments for the CURRENT block: issue early, consumed after softmax
    v16bf bv0 = load_frag(vbh, NKP,  0, kb0);
    v16bf bv1 = load_frag(vbh, NKP, 16, kb0);
    v16bf bv2 = load_frag(vbh, NKP, 32, kb0);
    v16bf bv3 = load_frag(vbh, NKP, 48, kb0);
    // K fragments for the NEXT block: in flight across the whole iteration
    if (kblk + 1 < nkb) {
      int kn = kb0 + 32;
      k00 = load_frag(kbh, D_, kn,      0);
      k01 = load_frag(kbh, D_, kn,      32);
      k10 = load_frag(kbh, D_, kn + 16, 0);
      k11 = load_frag(kbh, D_, kn + 16, 32);
    }

    // S(16x32) = Q(16x64) @ K^T : two 16x16 C tiles, two K-steps of 32
    v8f s0 = {}, s1 = {};
    s0 = wmma_bf16(aq0, ck00, s0); s0 = wmma_bf16(aq1, ck01, s0);
    s1 = wmma_bf16(aq0, ck10, s1); s1 = wmma_bf16(aq1, ck11, s1);

    bool ok0 = (kb0 + l15) < k_len;        // column validity per lane
    bool ok1 = (kb0 + 16 + l15) < k_len;

    float p0[8], p1[8];
    #pragma unroll
    for (int j = 0; j < 8; ++j) {
      p0[j] = ok0 ? s0[j] : -1e30f;
      p1[j] = ok1 ? s1[j] : -1e30f;
    }
    #pragma unroll
    for (int j = 0; j < 8; ++j) {
      float bm   = half_reduce_max(fmaxf(p0[j], p1[j]));
      float mnew = fmaxf(mrow[j], bm);
      float alpha = __expf(mrow[j] - mnew);
      mrow[j] = mnew;
      p0[j] = __expf(p0[j] - mnew);
      p1[j] = __expf(p1[j] - mnew);
      float rs = half_reduce_sum(p0[j] + p1[j]);
      lrow[j] = lrow[j] * alpha + rs;
      o0[j] *= alpha; o1[j] *= alpha; o2[j] *= alpha; o3[j] *= alpha;
      // C-layout element (row hi*8+j, col tile*16+l15) -> LDS (bf16)
      my[(hi * 8 + j) * 32 + l15]      = f2bf(p0[j]);
      my[(hi * 8 + j) * 32 + 16 + l15] = f2bf(p1[j]);
    }
    // Re-read P in A-layout (C->A transpose through LDS, per-wave slab)
    v16bf ap;
    const ushort_t* lp = my + l15 * 32;
    ((uint4*)&ap)[0] = *(const uint4*)(lp + hi * 8);
    ((uint4*)&ap)[1] = *(const uint4*)(lp + 16 + hi * 8);

    // O(16x64) += P(16x32) @ V(32x64); B-source = vT[D, NKP]
    o0 = wmma_bf16(ap, bv0, o0); o1 = wmma_bf16(ap, bv1, o1);
    o2 = wmma_bf16(ap, bv2, o2); o3 = wmma_bf16(ap, bv3, o3);
  }

  #pragma unroll
  for (int j = 0; j < 8; ++j) {
    int qrow = qbase + hi * 8 + j;
    if (qrow < q_len) {
      float inv = 1.0f / lrow[j];
      size_t row = (size_t)b * NTOK + q_start + qrow;
      ushort_t* orow = ao + row * C_ + h * D_;
      orow[l15]      = f2bf(o0[j] * inv);
      orow[16 + l15] = f2bf(o1[j] * inv);
      orow[32 + l15] = f2bf(o2[j] * inv);
      orow[48 + l15] = f2bf(o3[j] * inv);
    }
  }
}

// ---------- kernel 3: output projection (ao @ proj_w^T + b) -> f32 ----------
// grid (194, 6), block 256 (8 waves as 4M x 2N), wave tile 32M x 64N, pipelined.
__global__ __launch_bounds__(256) void proj_gemm_kernel(
    const ushort_t* __restrict__ ab, const ushort_t* __restrict__ wb,
    const float* __restrict__ bias, float* __restrict__ out) {
  int wid = threadIdx.x >> 5;
  int lane = threadIdx.x & 31, l15 = lane & 15, hi = lane >> 4;
  int mb = blockIdx.x * 128 + (wid >> 1) * 32;
  int nb = blockIdx.y * 128 + (wid & 1) * 64;

  v8f c[2][4] = {};
  v16bf a0 = load_frag(ab, C_, mb,      0);
  v16bf a1 = load_frag(ab, C_, mb + 16, 0);
  v16bf b0 = load_frag(wb, C_, nb,      0);
  v16bf b1 = load_frag(wb, C_, nb + 16, 0);
  v16bf b2 = load_frag(wb, C_, nb + 32, 0);
  v16bf b3 = load_frag(wb, C_, nb + 48, 0);

  for (int ks = 0; ks < C_; ks += 32) {
    v16bf ca0 = a0, ca1 = a1, cb0 = b0, cb1 = b1, cb2 = b2, cb3 = b3;
    int nk = ks + 32;
    if (nk < C_) {
      a0 = load_frag(ab, C_, mb,      nk);
      a1 = load_frag(ab, C_, mb + 16, nk);
      b0 = load_frag(wb, C_, nb,      nk);
      b1 = load_frag(wb, C_, nb + 16, nk);
      b2 = load_frag(wb, C_, nb + 32, nk);
      b3 = load_frag(wb, C_, nb + 48, nk);
    }
    c[0][0] = wmma_bf16(ca0, cb0, c[0][0]);
    c[0][1] = wmma_bf16(ca0, cb1, c[0][1]);
    c[0][2] = wmma_bf16(ca0, cb2, c[0][2]);
    c[0][3] = wmma_bf16(ca0, cb3, c[0][3]);
    c[1][0] = wmma_bf16(ca1, cb0, c[1][0]);
    c[1][1] = wmma_bf16(ca1, cb1, c[1][1]);
    c[1][2] = wmma_bf16(ca1, cb2, c[1][2]);
    c[1][3] = wmma_bf16(ca1, cb3, c[1][3]);
  }

  #pragma unroll
  for (int s = 0; s < 2; ++s) {
    #pragma unroll
    for (int t = 0; t < 4; ++t) {
      int col = nb + t * 16 + l15;
      float bv = bias[col];
      #pragma unroll
      for (int j = 0; j < 8; ++j) {
        size_t r = (size_t)mb + s * 16 + hi * 8 + j;
        out[r * C_ + col] = c[s][t][j] + bv;
      }
    }
  }
}

// ---------- host side ----------
extern "C" void kernel_launch(void* const* d_in, const int* in_sizes, int n_in,
                              void* d_out, int out_size, void* d_ws, size_t ws_size,
                              hipStream_t stream) {
  const float* x      = (const float*)d_in[0];
  const float* qkv_w  = (const float*)d_in[1];
  const float* qkv_b  = (const float*)d_in[2];
  const float* proj_w = (const float*)d_in[3];
  const float* proj_b = (const float*)d_in[4];
  float* out = (float*)d_out;

  char* ws = (char*)d_ws;
  size_t off = 0;
  auto alloc = [&](size_t bytes) -> char* {
    char* p = ws + off;
    off = (off + bytes + 255) & ~(size_t)255;
    return p;
  };
  ushort_t* x_bf    = (ushort_t*)alloc((size_t)MTOT * C_ * 2);
  ushort_t* qkvw_bf = (ushort_t*)alloc((size_t)3 * C_ * C_ * 2);
  ushort_t* projw_bf= (ushort_t*)alloc((size_t)C_ * C_ * 2);
  ushort_t* q_bf    = (ushort_t*)alloc((size_t)B_ * H_ * NKP * D_ * 2);
  ushort_t* k_bf    = (ushort_t*)alloc((size_t)B_ * H_ * NKP * D_ * 2);
  ushort_t* vT_bf   = (ushort_t*)alloc((size_t)B_ * H_ * D_ * NKP * 2);
  ushort_t* ao_bf   = (ushort_t*)alloc((size_t)MTOT * C_ * 2);

  // 0) bf16 conversions + padding init
  {
    int n4 = (MTOT * C_) / 4;
    cvt4_kernel<<<(n4 + 255) / 256, 256, 0, stream>>>((const float4*)x, (ushort4*)x_bf, n4);
    n4 = (3 * C_ * C_) / 4;
    cvt4_kernel<<<(n4 + 255) / 256, 256, 0, stream>>>((const float4*)qkv_w, (ushort4*)qkvw_bf, n4);
    n4 = (C_ * C_) / 4;
    cvt4_kernel<<<(n4 + 255) / 256, 256, 0, stream>>>((const float4*)proj_w, (ushort4*)projw_bf, n4);
    int npad = B_ * H_ * (NKP - NTOK) * D_;
    zero_pad_kernel<<<(npad + 255) / 256, 256, 0, stream>>>(q_bf, k_bf, vT_bf);
  }

  // 1) QKV projection GEMM -> q/k/vT bf16
  qkv_gemm_kernel<<<dim3(MTOT / 128, (3 * C_) / 128), 256, 0, stream>>>(
      x_bf, qkvw_bf, qkv_b, q_bf, k_bf, vT_bf);

  // 2) attention: temporal (q 0..127, k 0..127), spatial (q 128..387, k 0..387)
  flash_kernel<<<dim3((LT + 63) / 64, B_ * H_), 128, 0, stream>>>(
      q_bf, k_bf, vT_bf, ao_bf, /*q_start=*/0, /*q_len=*/LT, /*k_len=*/LT);
  flash_kernel<<<dim3((LS + 63) / 64, B_ * H_), 128, 0, stream>>>(
      q_bf, k_bf, vT_bf, ao_bf, /*q_start=*/LT, /*q_len=*/LS, /*k_len=*/NTOK);

  // 3) output projection -> f32
  proj_gemm_kernel<<<dim3(MTOT / 128, C_ / 128), 256, 0, stream>>>(
      ao_bf, projw_bf, proj_b, out);
}
